// ContextualAttentionLayer_22170621182179
// MI455X (gfx1250) — matrix operand order repaired and checked by reading, not compiled, not run on hardware
//
#include <hip/hip_runtime.h>
#include <hip/hip_bf16.h>

// Contextual attention (DeepFill) for MI455X / gfx1250.
// Heavy ops (two big GEMMs) run on v_wmma_f32_16x16x32_f16 with f32 accumulate,
// fed by CDNA5 async global->LDS DMA (global_load_async_to_lds_b128, ASYNCcnt)
// with double-buffered LDS tiles so global latency hides behind WMMA.
// Norm scaling, fuse stencils and softmax stay in f32 for accuracy.
// Workspace requirement: ~210 MB (see offsets in kernel_launch).

typedef _Float16 half_t;
typedef __attribute__((ext_vector_type(16))) _Float16 v16h;
typedef __attribute__((ext_vector_type(8)))  _Float16 v8h;
typedef __attribute__((ext_vector_type(8)))  float    v8f;

#define NB    4
#define HIMG  96
#define CCH   64
#define HD    48
#define LTOT  2304   // 48*48
#define K1    576    // 3*3*64
#define N2    1024   // 4*4*64

// ---------- build 3x3 patch matrix (f16) from downsampled grid (SAME pad) ----
// dst[b][l][(ky*3+kx)*64 + c] = src_down[b, li+ky-1, lj+kx-1, c]  (src_down = src[:,1::2,1::2,:])
__global__ __launch_bounds__(256) void build_patch3(const float* __restrict__ src,
                                                    half_t* __restrict__ dst)
{
    int idx = blockIdx.x * 256 + threadIdx.x;
    int c  = idx & 63;
    int r  = idx >> 6;
    int kx = r % 3; r /= 3;
    int ky = r % 3; r /= 3;
    int l  = r % LTOT;
    int b  = r / LTOT;
    int li = l / HD, lj = l % HD;
    int row = li + ky - 1, col = lj + kx - 1;
    float v = 0.f;
    if (row >= 0 && row < HD && col >= 0 && col < HD)
        v = src[(((b * HIMG) + (2*row+1)) * HIMG + (2*col+1)) * CCH + c];
    dst[idx] = (half_t)v;
}

// ---------- build raw_w as [N2 x L] per sample (4x4 patches of b, stride 2, pad 1) ----
__global__ __launch_bounds__(256) void build_rw(const float* __restrict__ bsrc,
                                                half_t* __restrict__ dst)
{
    int idx = blockIdx.x * 256 + threadIdx.x;
    int l = idx % LTOT;
    int r = idx / LTOT;
    int c  = r & 63; r >>= 6;
    int kx = r & 3;  r >>= 2;
    int ky = r & 3;  r >>= 2;
    int b  = r;
    int li = l / HD, lj = l % HD;
    int row = 2*li - 1 + ky, col = 2*lj - 1 + kx;
    float v = 0.f;
    if (row >= 0 && row < HIMG && col >= 0 && col < HIMG)
        v = bsrc[((b * HIMG + row) * HIMG + col) * CCH + c];
    dst[idx] = (half_t)v;
}

// ---------- per-filter 1/max(||w||,1e-4), computed in f32 from source ----
__global__ __launch_bounds__(256) void rnorm_k(const float* __restrict__ bsrc,
                                               float* __restrict__ rnorm)
{
    int gid  = blockIdx.x * 8 + (threadIdx.x >> 5);   // one wave32 per (b,l)
    int lane = threadIdx.x & 31;
    int b = gid / LTOT, l = gid % LTOT;
    int li = l / HD, lj = l % HD;
    float s = 0.f;
    #pragma unroll
    for (int i = 0; i < 18; ++i) {
        int k = lane + 32 * i;                        // k < 576
        int c = k & 63, kk = k >> 6;
        int kx = kk % 3, ky = kk / 3;
        int row = li + ky - 1, col = lj + kx - 1;
        if (row >= 0 && row < HD && col >= 0 && col < HD) {
            float v = bsrc[(((b * HIMG) + (2*row+1)) * HIMG + (2*col+1)) * CCH + c];
            s += v * v;
        }
    }
    #pragma unroll
    for (int off = 16; off; off >>= 1) s += __shfl_xor(s, off, 32);
    if (lane == 0) rnorm[gid] = 1.f / fmaxf(sqrtf(s), 1e-4f);
}

// ---------- mask gate: mm[b][l] = (mean of 3x3 patch of downsampled mask == 0) ----
__global__ void mm_k(const float* __restrict__ mask, float* __restrict__ mm)
{
    int gid = blockIdx.x * blockDim.x + threadIdx.x;
    if (gid >= NB * LTOT) return;
    int b = gid / LTOT, l = gid % LTOT;
    int li = l / HD, lj = l % HD;
    float s = 0.f;
    for (int ky = 0; ky < 3; ++ky)
        for (int kx = 0; kx < 3; ++kx) {
            int row = li + ky - 1, col = lj + kx - 1;
            if (row >= 0 && row < HD && col >= 0 && col < HD)
                s += mask[(b * HIMG + (2*row+1)) * HIMG + (2*col+1)];
        }
    mm[gid] = (s * (1.f / 9.f) == 0.f) ? 1.f : 0.f;
}

// ---------- CDNA5 async global->LDS 16B copy (per lane), ASYNCcnt tracked ----
__device__ inline void async_cp16(uint32_t lds_addr, const half_t* g)
{
    asm volatile("global_load_async_to_lds_b128 %0, %1, off"
                 :: "v"(lds_addr), "v"(g) : "memory");
}

// ---------- WMMA GEMM: C[M,N] = A[M,K] * BT[N,K]^T, f16 in / f32 acc ----
// Block tile 128x64, BK=32, 8 waves in 4x2 grid, each wave computes 32x32
// (2 A-frags x 2 B-frags -> 4 v_wmma per K-step). Double-buffered LDS fed by
// async DMA with distance-1 prefetch; last tile peeled so the steady-state
// loop is branch-free: 3x DMA -> s_wait_asynccnt 3 -> barrier -> 8x ds_load
// -> 4x wmma -> barrier.
__device__ inline v16h join8(v8h lo, v8h hi)
{
    return __builtin_shufflevector(lo, hi, 0,1,2,3,4,5,6,7,8,9,10,11,12,13,14,15);
}

#define LDAP 40                    // padded LDS row stride (halves): 80B, conflict-free
#define ABUF (128 * LDAP)          // halves per A buffer
#define BBUF (64 * LDAP)           // halves per B buffer

template <int MODE>   // 0: f32 out scaled by colScale[n]; 1: f16 out
__global__ __launch_bounds__(256) void gemm_wmma(
    const half_t* __restrict__ A,    // per sample [M,K]
    const half_t* __restrict__ BT,   // per sample [N,K]
    float* __restrict__ Cf, half_t* __restrict__ Ch,
    const float* __restrict__ colScale,
    int M, int N, int K)
{
    __shared__ __align__(16) half_t Asm[2 * ABUF];   // 20.0 KB
    __shared__ __align__(16) half_t Bsm[2 * BBUF];   // 10.0 KB

    const int bz = blockIdx.z;
    A  += (size_t)bz * M * K;
    BT += (size_t)bz * N * K;
    if (MODE == 0) { Cf += (size_t)bz * M * N; colScale += (size_t)bz * N; }
    else           { Ch += (size_t)bz * M * N; }

    const int tid  = threadIdx.x;
    const int lane = tid & 31, wave = tid >> 5;
    const int wr = wave & 3, wc = wave >> 2;          // 4x2 wave grid over 128x64 tile
    const int m0 = blockIdx.y * 128, n0 = blockIdx.x * 64;

    // async-copy source/dest for this thread (A: 2 x 16B per tile, B: 1 x 16B)
    const half_t* gA0 = A  + (size_t)(m0 + (tid >> 1)) * K + (tid & 1) * 16;
    const half_t* gB0 = BT + (size_t)(n0 + (tid >> 2)) * K + (tid & 3) * 8;
    const uint32_t ldsA = (uint32_t)(size_t)Asm + ((tid >> 1) * LDAP + (tid & 1) * 16) * 2;
    const uint32_t ldsB = (uint32_t)(size_t)Bsm + ((tid >> 2) * LDAP + (tid & 3) * 8) * 2;

    // fragment addressing per CDNA5 ISA 7.12.2
    const int fm    = lane & 15;
    const int khalf = (lane >> 4) * 8;                // A frag: K-block select
    const int kb    = (lane >> 4) * 16;               // unused: B frag also uses khalf

    v8f acc00 = {}, acc01 = {}, acc10 = {}, acc11 = {};

    auto compute_tile = [&](int buf) {
        const half_t* aB = Asm + buf * ABUF;
        const half_t* bB = Bsm + buf * BBUF;
        const half_t* aRow0 = aB + (wr * 32 + fm) * LDAP;
        const half_t* aRow1 = aB + (wr * 32 + 16 + fm) * LDAP;
        const half_t* bRow0 = bB + (wc * 32 + fm) * LDAP;
        const half_t* bRow1 = bB + (wc * 32 + 16 + fm) * LDAP;

        v16h a0 = join8(*(const v8h*)(aRow0 + khalf), *(const v8h*)(aRow0 + 16 + khalf));
        v16h a1 = join8(*(const v8h*)(aRow1 + khalf), *(const v8h*)(aRow1 + 16 + khalf));
        v16h b0 = join8(*(const v8h*)(bRow0 + khalf), *(const v8h*)(bRow0 + 16 + khalf));
        v16h b1 = join8(*(const v8h*)(bRow1 + khalf), *(const v8h*)(bRow1 + 16 + khalf));

        acc00 = __builtin_amdgcn_wmma_f32_16x16x32_f16(false, a0, false, b0, (short)0, acc00, false, false);
        acc01 = __builtin_amdgcn_wmma_f32_16x16x32_f16(false, a0, false, b1, (short)0, acc01, false, false);
        acc10 = __builtin_amdgcn_wmma_f32_16x16x32_f16(false, a1, false, b0, (short)0, acc10, false, false);
        acc11 = __builtin_amdgcn_wmma_f32_16x16x32_f16(false, a1, false, b1, (short)0, acc11, false, false);
    };

    // prologue: tile 0 -> buffer 0 (ASYNCcnt = 3)
    async_cp16(ldsA,      gA0);
    async_cp16(ldsA + 16, gA0 + 8);
    async_cp16(ldsB,      gB0);

    const int nk = K >> 5;
    // steady state: branch-free — always prefetch next tile, wait for current
    for (int kt = 0; kt + 1 < nk; ++kt) {
        const int buf = kt & 1;
        const half_t* ga = gA0 + ((kt + 1) << 5);
        const half_t* gb = gB0 + ((kt + 1) << 5);
        const uint32_t la = ldsA + (buf ^ 1) * (ABUF * 2);
        const uint32_t lb = ldsB + (buf ^ 1) * (BBUF * 2);
        async_cp16(la,      ga);
        async_cp16(la + 16, ga + 8);
        async_cp16(lb,      gb);
        asm volatile("s_wait_asynccnt 0x3" ::: "memory");   // in-order: tile kt landed
        __syncthreads();                                    // all waves' DMA visible
        compute_tile(buf);
        __syncthreads();            // all frag reads done before next prefetch overwrites
    }

    // epilogue: last tile, no prefetch
    asm volatile("s_wait_asynccnt 0x0" ::: "memory");
    __syncthreads();
    compute_tile((nk - 1) & 1);

    const int or0 = m0 + wr * 32 + (lane >> 4) * 8;   // rows for acc*0 vgpr r
    const int oc0 = n0 + wc * 32 + fm;
    const int oc1 = oc0 + 16;
    if (MODE == 0) {
        float s0 = colScale[oc0], s1 = colScale[oc1];
        #pragma unroll
        for (int r2 = 0; r2 < 8; ++r2) {
            Cf[(size_t)(or0 + r2) * N + oc0]      = acc00[r2] * s0;
            Cf[(size_t)(or0 + r2) * N + oc1]      = acc01[r2] * s1;
            Cf[(size_t)(or0 + 16 + r2) * N + oc0] = acc10[r2] * s0;
            Cf[(size_t)(or0 + 16 + r2) * N + oc1] = acc11[r2] * s1;
        }
    } else {
        #pragma unroll
        for (int r2 = 0; r2 < 8; ++r2) {
            Ch[(size_t)(or0 + r2) * N + oc0]      = (half_t)acc00[r2];
            Ch[(size_t)(or0 + r2) * N + oc1]      = (half_t)acc01[r2];
            Ch[(size_t)(or0 + 16 + r2) * N + oc0] = (half_t)acc10[r2];
            Ch[(size_t)(or0 + 16 + r2) * N + oc1] = (half_t)acc11[r2];
        }
    }
}

// ---------- 3x3 fuse stencil on LxL map, writing through the (a,b)x(c,d) swap ----
__global__ __launch_bounds__(256) void fuse_perm(const float* __restrict__ in,
                                                 float* __restrict__ out,
                                                 const float* __restrict__ fw)
{
    int idx = blockIdx.x * 256 + threadIdx.x;
    int l = idx % LTOT;
    int r = idx / LTOT;
    int p = r % LTOT;
    int b = r / LTOT;
    const float* base = in + (size_t)b * LTOT * LTOT;
    float acc = 0.f;
    #pragma unroll
    for (int i = 0; i < 3; ++i) {
        int pp = p + i - 1;
        if (pp < 0 || pp >= LTOT) continue;
        #pragma unroll
        for (int j = 0; j < 3; ++j) {
            int ll = l + j - 1;
            if (ll < 0 || ll >= LTOT) continue;
            acc += fw[i * 3 + j] * base[(size_t)pp * LTOT + ll];
        }
    }
    int a  = p / HD, bb = p % HD;
    int c0 = l / HD, d  = l % HD;
    out[(size_t)b * LTOT * LTOT + (size_t)(bb * HD + a) * LTOT + (d * HD + c0)] = acc;
}

// ---------- masked softmax over rows (scale 10), f16 output ----
__global__ __launch_bounds__(256) void softmax_k(const float* __restrict__ Y,
                                                 const float* __restrict__ mm,
                                                 half_t* __restrict__ S)
{
    __shared__ float red[256];
    int row = blockIdx.x;                 // (b,p)
    int b = row / LTOT;
    const float* in   = Y  + (size_t)row * LTOT;
    const float* mrow = mm + (size_t)b * LTOT;
    half_t* out = S + (size_t)row * LTOT;
    int t = threadIdx.x;

    float v[9]; float lmax = -3.4e38f;
    #pragma unroll
    for (int i = 0; i < 9; ++i) {
        int l = t + 256 * i;
        float x = in[l] * mrow[l];
        v[i] = x;
        lmax = fmaxf(lmax, x);
    }
    red[t] = lmax; __syncthreads();
    for (int s2 = 128; s2 > 0; s2 >>= 1) { if (t < s2) red[t] = fmaxf(red[t], red[t + s2]); __syncthreads(); }
    float xmax = red[0]; __syncthreads();

    float e[9]; float lsum = 0.f;
    #pragma unroll
    for (int i = 0; i < 9; ++i) { e[i] = __expf(10.f * (v[i] - xmax)); lsum += e[i]; }
    red[t] = lsum; __syncthreads();
    for (int s2 = 128; s2 > 0; s2 >>= 1) { if (t < s2) red[t] += red[t + s2]; __syncthreads(); }
    float inv = 1.f / red[0];

    #pragma unroll
    for (int i = 0; i < 9; ++i) {
        int l = t + 256 * i;
        out[l] = (half_t)(e[i] * inv * mrow[l]);
    }
}

// ---------- transpose-conv gather: out[b,i,j,c] = 0.25 * sum over <=2x2 taps of G ----
__global__ __launch_bounds__(256) void gather_k(const half_t* __restrict__ G,
                                                float* __restrict__ out)
{
    int idx = blockIdx.x * 256 + threadIdx.x;
    int c = idx & 63;
    int r = idx >> 6;
    int j = r % HIMG; r /= HIMG;
    int i = r % HIMG;
    int b = r / HIMG;
    float acc = 0.f;
    #pragma unroll
    for (int ky = 0; ky < 4; ++ky) {
        int ii = i + 1 - ky;
        if (ii < 0 || (ii & 1)) continue;
        int pi = ii >> 1; if (pi >= HD) continue;
        #pragma unroll
        for (int kx = 0; kx < 4; ++kx) {
            int jj = j + 1 - kx;
            if (jj < 0 || (jj & 1)) continue;
            int pj = jj >> 1; if (pj >= HD) continue;
            acc += (float)G[((size_t)b * LTOT + pi * HD + pj) * N2 + (ky * 4 + kx) * 64 + c];
        }
    }
    out[idx] = acc * 0.25f;
}

extern "C" void kernel_launch(void* const* d_in, const int* in_sizes, int n_in,
                              void* d_out, int out_size, void* d_ws, size_t ws_size,
                              hipStream_t stream)
{
    const float* f    = (const float*)d_in[0];   // (4,96,96,64)
    const float* bsrc = (const float*)d_in[1];   // (4,96,96,64)
    const float* mask = (const float*)d_in[2];   // (4,96,96,1)
    const float* fw   = (const float*)d_in[3];   // (3,3,1,1)
    float* out = (float*)d_out;                  // (4,96,96,64)

    char* ws = (char*)d_ws;
    size_t off = 0;
    half_t* Abuf  = (half_t*)(ws + off); off += (size_t)NB * LTOT * K1 * sizeof(half_t);   // 10.6 MB
    half_t* Bbuf  = (half_t*)(ws + off); off += (size_t)NB * LTOT * K1 * sizeof(half_t);   // 10.6 MB
    half_t* RWbuf = (half_t*)(ws + off); off += (size_t)NB * LTOT * N2 * sizeof(half_t);   // 18.9 MB
    float*  rnorm = (float*)(ws + off);  off += (size_t)NB * LTOT * sizeof(float);
    float*  mmbuf = (float*)(ws + off);  off += (size_t)NB * LTOT * sizeof(float);
    float*  Ybuf  = (float*)(ws + off);  off += (size_t)NB * LTOT * LTOT * sizeof(float);  // 84.9 MB
    float*  Zbuf  = (float*)(ws + off);  off += (size_t)NB * LTOT * LTOT * sizeof(float);  // 84.9 MB
    half_t* Sbuf  = (half_t*)Zbuf;   // reuse: Z dead after second fuse pass (needs 42.5 MB)
    half_t* Gbuf  = (half_t*)Abuf;   // reuse: A+Bn (21.2 MB) dead after GEMM1; G needs 18.9 MB
    (void)in_sizes; (void)n_in; (void)out_size; (void)ws_size;

    // ---- patch/weight construction ----
    build_patch3<<<NB * LTOT * K1 / 256, 256, 0, stream>>>(f,    Abuf);
    build_patch3<<<NB * LTOT * K1 / 256, 256, 0, stream>>>(bsrc, Bbuf);
    build_rw    <<<NB * LTOT * N2 / 256, 256, 0, stream>>>(bsrc, RWbuf);
    rnorm_k     <<<NB * LTOT / 8,        256, 0, stream>>>(bsrc, rnorm);
    mm_k        <<<(NB * LTOT + 255) / 256, 256, 0, stream>>>(mask, mmbuf);

    // ---- GEMM1: Y[p,l] = (A . B[l]) * rnorm[l]   (M=N=2304, K=576) ----
    gemm_wmma<0><<<dim3(LTOT / 64, LTOT / 128, NB), 256, 0, stream>>>(
        Abuf, Bbuf, Ybuf, nullptr, rnorm, LTOT, LTOT, K1);

    // ---- fuse conv x2 with interleaved index swap ----
    fuse_perm<<<NB * LTOT * LTOT / 256, 256, 0, stream>>>(Ybuf, Zbuf, fw);
    fuse_perm<<<NB * LTOT * LTOT / 256, 256, 0, stream>>>(Zbuf, Ybuf, fw);

    // ---- masked softmax (scale 10) -> f16 S ----
    softmax_k<<<NB * LTOT, 256, 0, stream>>>(Ybuf, mmbuf, Sbuf);

    // ---- GEMM2: G[p,n] = S[p,:] . RW[n,:]   (M=2304, N=1024, K=2304) ----
    gemm_wmma<1><<<dim3(N2 / 64, LTOT / 128, NB), 256, 0, stream>>>(
        Sbuf, RWbuf, nullptr, Gbuf, nullptr, LTOT, N2, LTOT);

    // ---- transpose-conv gather into output ----
    gather_k<<<NB * HIMG * HIMG * CCH / 256, 256, 0, stream>>>(Gbuf, out);
}
